// GraphRNNDecoder_18872086298991
// MI455X (gfx1250) — compile-verified, matrix operand order checked by reading, not run on hardware
//
#include <hip/hip_runtime.h>
#include <stdint.h>

#define Hd   768
#define Bsz  32
#define NL   4
#define MAXN 12
#define BH   (Bsz*Hd)                    // 24576 floats
#define LBH  (NL*BH)                     // 98304 floats
#define PK_PER_LAYER  (144*24*32*8)      // 884736 u32 per (layer, weight matrix)
#define PK_PER_TENSOR ((size_t)NL*PK_PER_LAYER)
#define AP_LAYER      (2*24*32*8)        // 12288 u32: one activation (32x768) in A-frag order
#define AP_TENSOR     (NL*AP_LAYER)      // 49152 u32 per activation stack buffer

typedef __attribute__((ext_vector_type(16))) __bf16         v16bf;
typedef __attribute__((ext_vector_type(16))) unsigned short v16u;
typedef __attribute__((ext_vector_type(8)))  float          v8f;
typedef __attribute__((ext_vector_type(8)))  unsigned int   v8u;

__device__ __forceinline__ unsigned short f2bf(float f) {
  unsigned int u = __float_as_uint(f);
  u += 0x7FFFu + ((u >> 16) & 1u);          // round-to-nearest-even
  return (unsigned short)(u >> 16);
}

// bf16 fragment from fp32 memory (fallback path): elems 0..7 = p[0..7], 8..15 = p[16..23]
__device__ __forceinline__ v16bf frag_from_f32(const float* __restrict__ p) {
  const float4* q = (const float4*)p;
  float4 f0 = q[0], f1 = q[1], f2 = q[4], f3 = q[5];
  v16u v;
  v[0]=f2bf(f0.x);  v[1]=f2bf(f0.y);  v[2]=f2bf(f0.z);  v[3]=f2bf(f0.w);
  v[4]=f2bf(f1.x);  v[5]=f2bf(f1.y);  v[6]=f2bf(f1.z);  v[7]=f2bf(f1.w);
  v[8]=f2bf(f2.x);  v[9]=f2bf(f2.y);  v[10]=f2bf(f2.z); v[11]=f2bf(f2.w);
  v[12]=f2bf(f3.x); v[13]=f2bf(f3.y); v[14]=f2bf(f3.z); v[15]=f2bf(f3.w);
  return __builtin_bit_cast(v16bf, v);
}

// Pre-packed bf16 fragment: 8 consecutive dwords per lane -> two b128 loads.
__device__ __forceinline__ v16bf frag_from_pk(const uint32_t* __restrict__ p) {
  const uint4* q = (const uint4*)p;
  uint4 a = q[0], b = q[1];
  v8u w;
  w[0]=a.x; w[1]=a.y; w[2]=a.z; w[3]=a.w;
  w[4]=b.x; w[5]=b.y; w[6]=b.z; w[7]=b.w;
  return __builtin_bit_cast(v16bf, w);
}

__global__ void zero_kernel(float* __restrict__ p, int n) {
  int i = blockIdx.x * blockDim.x + threadIdx.x;
  if (i < n) p[i] = 0.0f;
}

// Pack weight tensor (L,3H,H) fp32 -> bf16 pairs in WMMA-B fragment order:
// [l][t=g*48+ntile][kt][lane][vi]; u32 = {bf16 W[row][k], bf16 W[row][k+1]}
__global__ void pack_weights(const float* __restrict__ W, uint32_t* __restrict__ out) {
  int idx = blockIdx.x * blockDim.x + threadIdx.x;
  if (idx >= NL * PK_PER_LAYER) return;
  int vi   = idx & 7;
  int lane = (idx >> 3) & 31;
  int kt   = (idx >> 8) % 24;
  int t    = ((idx >> 8) / 24) % 144;
  int l    = idx / PK_PER_LAYER;
  int g = t / 48, ntile = t % 48;
  int n  = lane & 15;
  int kb = (lane >> 4) << 3;
  int k  = kt * 32 + kb + (vi < 4 ? 2 * vi : 16 + 2 * (vi - 4));
  int row = l * (3 * Hd) + g * Hd + ntile * 16 + n;
  const float* src = W + (size_t)row * Hd + k;
  out[idx] = (uint32_t)f2bf(src[0]) | ((uint32_t)f2bf(src[1]) << 16);
}

// Pack one (32x768) fp32 activation -> A-fragment order: [mtile][kt][lane][vi]
__global__ void pack_act(const float* __restrict__ X, uint32_t* __restrict__ out) {
  int idx = blockIdx.x * blockDim.x + threadIdx.x;
  if (idx >= AP_LAYER) return;
  int vi    = idx & 7;
  int lane  = (idx >> 3) & 31;
  int kt    = (idx >> 8) % 24;
  int mtile = (idx >> 8) / 24;
  int kb = (lane >> 4) << 3;
  int k  = kt * 32 + kb + (vi < 4 ? 2 * vi : 16 + 2 * (vi - 4));
  int row = mtile * 16 + (lane & 15);
  const float* src = X + (size_t)row * Hd + k;
  out[idx] = (uint32_t)f2bf(src[0]) | ((uint32_t)f2bf(src[1]) << 16);
}

// ---------------- packed-path GRU layer-step ----------------
// Grid: 48 blocks x 64 threads. Block = 16-column slice; wave = 16-row M-tile.
// Reads x and h as pre-packed bf16 A-fragments; writes new h as fp32 AND packed bf16.
__launch_bounds__(64)
__global__ void gru_layer_pk(const uint32_t* __restrict__ xPk, const uint32_t* __restrict__ hPk,
                             const float* __restrict__ h_in, float* __restrict__ h_out,
                             uint32_t* __restrict__ outPk,
                             const uint32_t* __restrict__ WiP, const uint32_t* __restrict__ WhP,
                             const float* __restrict__ b_ih, const float* __restrict__ b_hh) {
  const int ntile = blockIdx.x;           // 0..47
  const int lane  = threadIdx.x & 31;
  const int wave  = threadIdx.x >> 5;     // 0..1 -> M-tile
  const int n15   = lane & 15;
  const int ncol  = (ntile << 4) + n15;   // output column within H (per gate)

  v8f acc[6];                             // 0..2: gi(r,z,n)  3..5: gh(r,z,n)
  #pragma unroll
  for (int q = 0; q < 6; ++q)
    #pragma unroll
    for (int e = 0; e < 8; ++e) acc[q][e] = 0.0f;

  for (int kt = 0; kt < 24; ++kt) {
    const uint32_t aoff = (uint32_t)(((wave * 24 + kt) * 32 + lane) * 8);
    v16bf ax = frag_from_pk(xPk + aoff);
    v16bf ah = frag_from_pk(hPk + aoff);
    #pragma unroll
    for (int g = 0; g < 3; ++g) {
      const uint32_t off = (uint32_t)((((g * 48 + ntile) * 24 + kt) * 32 + lane) * 8);
      v16bf bi = frag_from_pk(WiP + off);
      v16bf bh = frag_from_pk(WhP + off);
      if (kt < 23) {                      // next K-tile block is +256 dwords per lane
        __builtin_prefetch(WiP + off + 256, 0, 1);
        __builtin_prefetch(WhP + off + 256, 0, 1);
      }
      acc[g]     = __builtin_amdgcn_wmma_f32_16x16x32_bf16(false, ax, false, bi,
                                                           (short)0, acc[g], false, false);
      acc[3 + g] = __builtin_amdgcn_wmma_f32_16x16x32_bf16(false, ah, false, bh,
                                                           (short)0, acc[3 + g], false, false);
    }
  }

  // Gate math on C/D fragment: VGPR v -> M = v + 8*(lane>=16), N = lane&15.
  const float bir = b_ih[ncol], biz = b_ih[Hd + ncol], bin = b_ih[2 * Hd + ncol];
  const float bhr = b_hh[ncol], bhz = b_hh[Hd + ncol], bhn = b_hh[2 * Hd + ncol];
  const int hi8   = (lane >> 4) << 3;
  const int mbase = (wave << 4) + hi8;
  // packed-store constants: where column ncol lives inside the A-fragment layout
  const int kt_c = ncol >> 5;
  const int kk   = ncol & 31;
  const int ahl  = (kk >> 3) & 1;                      // lane-half select
  const int e16  = (kk & 7) + ((kk >> 4) << 3);        // u16 slot within lane block
  unsigned short* outU16 = (unsigned short*)outPk;
  #pragma unroll
  for (int v = 0; v < 8; ++v) {
    const int m = mbase + v;
    float r  = 1.0f / (1.0f + expf(-(acc[0][v] + bir + acc[3][v] + bhr)));
    float z  = 1.0f / (1.0f + expf(-(acc[1][v] + biz + acc[4][v] + bhz)));
    float nn = tanhf(acc[2][v] + bin + r * (acc[5][v] + bhn));
    float ho = h_in[m * Hd + ncol];
    float hn = (1.0f - z) * nn + z * ho;
    h_out[m * Hd + ncol] = hn;
    const int lane_a = hi8 + v + (ahl << 4);
    outU16[(((wave * 24 + kt_c) * 32 + lane_a) * 8) * 2 + e16] = f2bf(hn);
  }
}

// ---------------- fallback (unpacked fp32) GRU layer-step ----------------
__launch_bounds__(64)
__global__ void gru_layer_f32(const float* __restrict__ x, const float* __restrict__ h_in,
                              float* __restrict__ h_out,
                              const float* __restrict__ WiF, const float* __restrict__ WhF,
                              const float* __restrict__ b_ih, const float* __restrict__ b_hh) {
  const int ntile = blockIdx.x;
  const int lane  = threadIdx.x & 31;
  const int wave  = threadIdx.x >> 5;
  const int n15   = lane & 15;
  const int kb    = (lane >> 4) << 3;
  const int arow  = (wave << 4) + n15;
  const int ncol  = (ntile << 4) + n15;

  v8f acc[6];
  #pragma unroll
  for (int q = 0; q < 6; ++q)
    #pragma unroll
    for (int e = 0; e < 8; ++e) acc[q][e] = 0.0f;

  const float* xr = x + arow * Hd;
  const float* hr = h_in + arow * Hd;

  for (int kt = 0; kt < 24; ++kt) {
    const int k0 = kt * 32 + kb;
    v16bf ax = frag_from_f32(xr + k0);
    v16bf ah = frag_from_f32(hr + k0);
    #pragma unroll
    for (int g = 0; g < 3; ++g) {
      const size_t row = (size_t)(g * Hd + ncol);
      v16bf bi = frag_from_f32(WiF + row * Hd + k0);
      v16bf bh = frag_from_f32(WhF + row * Hd + k0);
      acc[g]     = __builtin_amdgcn_wmma_f32_16x16x32_bf16(false, ax, false, bi,
                                                           (short)0, acc[g], false, false);
      acc[3 + g] = __builtin_amdgcn_wmma_f32_16x16x32_bf16(false, ah, false, bh,
                                                           (short)0, acc[3 + g], false, false);
    }
  }

  const float bir = b_ih[ncol], biz = b_ih[Hd + ncol], bin = b_ih[2 * Hd + ncol];
  const float bhr = b_hh[ncol], bhz = b_hh[Hd + ncol], bhn = b_hh[2 * Hd + ncol];
  const int mbase = (wave << 4) + ((lane >> 4) << 3);
  #pragma unroll
  for (int v = 0; v < 8; ++v) {
    const int m = mbase + v;
    float r  = 1.0f / (1.0f + expf(-(acc[0][v] + bir + acc[3][v] + bhr)));
    float z  = 1.0f / (1.0f + expf(-(acc[1][v] + biz + acc[4][v] + bhz)));
    float nn = tanhf(acc[2][v] + bin + r * (acc[5][v] + bhn));
    float ho = h_in[m * Hd + ncol];
    h_out[m * Hd + ncol] = (1.0f - z) * nn + z * ho;
  }
}

// sigmoid(inp @ W^T + b); one wave, lane = batch row; writes dst[b*MAXN].
__global__ void head_sigmoid(const float* __restrict__ inp, const float* __restrict__ W,
                             const float* __restrict__ bias, float* __restrict__ dst) {
  int b = threadIdx.x;
  if (b >= Bsz) return;
  const float4* o = (const float4*)(inp + b * Hd);
  const float4* w = (const float4*)W;
  float s = 0.0f;
  for (int k = 0; k < Hd / 4; ++k) {
    float4 a = o[k], c = w[k];
    s += a.x * c.x + a.y * c.y + a.z * c.z + a.w * c.w;
  }
  s += bias[0];
  dst[b * MAXN] = 1.0f / (1.0f + expf(-s));
}

extern "C" void kernel_launch(void* const* d_in, const int* in_sizes, int n_in,
                              void* d_out, int out_size, void* d_ws, size_t ws_size,
                              hipStream_t stream) {
  const float* ge    = (const float*)d_in[0];
  const float* Wg_ih = (const float*)d_in[1];
  const float* Wg_hh = (const float*)d_in[2];
  const float* bg_ih = (const float*)d_in[3];
  const float* bg_hh = (const float*)d_in[4];
  const float* We_ih = (const float*)d_in[5];
  const float* We_hh = (const float*)d_in[6];
  const float* be_ih = (const float*)d_in[7];
  const float* be_hh = (const float*)d_in[8];
  const float* Wn    = (const float*)d_in[9];
  const float* bn    = (const float*)d_in[10];
  const float* Wf    = (const float*)d_in[11];
  const float* bf    = (const float*)d_in[12];
  float* out = (float*)d_out;

  // ws layout: fp32 states | packed activations | packed weights
  float* ws = (float*)d_ws;
  float* hg[2] = { ws, ws + LBH };
  float* he[2] = { ws + 2 * LBH, ws + 3 * LBH };
  uint32_t* actPk = (uint32_t*)(ws + 4 * LBH);
  uint32_t* hgPk[2] = { actPk, actPk + AP_TENSOR };
  uint32_t* hePk[2] = { actPk + 2 * AP_TENSOR, actPk + 3 * AP_TENSOR };
  uint32_t* gePk = actPk + 4 * AP_TENSOR;
  uint32_t* pkW  = gePk + AP_LAYER;
  uint32_t* pWgI = pkW;
  uint32_t* pWgH = pkW + PK_PER_TENSOR;
  uint32_t* pWeI = pkW + 2 * PK_PER_TENSOR;
  uint32_t* pWeH = pkW + 3 * PK_PER_TENSOR;
  const size_t needBytes = (size_t)4 * LBH * 4 + (size_t)(4 * AP_TENSOR + AP_LAYER) * 4 +
                           4 * PK_PER_TENSOR * 4;
  const bool packed = ws_size >= needBytes;

  // Zero fp32 states + packed activation states (h0 = 0), and the output tensor.
  const int zeroN = 4 * LBH + 4 * AP_TENSOR + AP_LAYER;
  zero_kernel<<<(zeroN + 255) / 256, 256, 0, stream>>>(ws, zeroN);
  zero_kernel<<<(Bsz * MAXN * (1 + MAXN) + 255) / 256, 256, 0, stream>>>(
      out, Bsz * MAXN * (1 + MAXN));

  if (packed) {
    const int pkBlocks = (NL * PK_PER_LAYER + 255) / 256;
    pack_weights<<<pkBlocks, 256, 0, stream>>>(Wg_ih, pWgI);
    pack_weights<<<pkBlocks, 256, 0, stream>>>(Wg_hh, pWgH);
    pack_weights<<<pkBlocks, 256, 0, stream>>>(We_ih, pWeI);
    pack_weights<<<pkBlocks, 256, 0, stream>>>(We_hh, pWeH);
    pack_act<<<(AP_LAYER + 255) / 256, 256, 0, stream>>>(ge, gePk);
  }

  int pg = 0, pe = 0;
  for (int i = 0; i < MAXN; ++i) {
    float* src = hg[pg];
    float* dst = hg[pg ^ 1];
    uint32_t* srcPk = hgPk[pg];
    uint32_t* dstPk = hgPk[pg ^ 1];
    pg ^= 1;
    const float*    xin  = ge;
    const uint32_t* xinP = gePk;
    for (int l = 0; l < NL; ++l) {
      if (packed)
        gru_layer_pk<<<48, 64, 0, stream>>>(
            xinP, srcPk + l * AP_LAYER, src + l * BH, dst + l * BH, dstPk + l * AP_LAYER,
            pWgI + (size_t)l * PK_PER_LAYER, pWgH + (size_t)l * PK_PER_LAYER,
            bg_ih + l * 3 * Hd, bg_hh + l * 3 * Hd);
      else
        gru_layer_f32<<<48, 64, 0, stream>>>(
            xin, src + l * BH, dst + l * BH,
            Wg_ih + (size_t)l * 3 * Hd * Hd, Wg_hh + (size_t)l * 3 * Hd * Hd,
            bg_ih + l * 3 * Hd, bg_hh + l * 3 * Hd);
      xin  = dst + l * BH;
      xinP = dstPk + l * AP_LAYER;
    }
    const float*    out_g  = dst + 3 * BH;
    const uint32_t* out_gP = dstPk + 3 * AP_LAYER;

    head_sigmoid<<<1, 32, 0, stream>>>(out_g, Wn, bn, out + i);   // nodes[b][i]

    for (int j = 0; j <= i; ++j) {
      float* esrc = he[pe];
      float* edst = he[pe ^ 1];
      uint32_t* esrcPk = hePk[pe];
      uint32_t* edstPk = hePk[pe ^ 1];
      pe ^= 1;
      const float*    exin  = out_g;
      const uint32_t* exinP = out_gP;
      for (int l = 0; l < NL; ++l) {
        if (packed)
          gru_layer_pk<<<48, 64, 0, stream>>>(
              exinP, esrcPk + l * AP_LAYER, esrc + l * BH, edst + l * BH,
              edstPk + l * AP_LAYER,
              pWeI + (size_t)l * PK_PER_LAYER, pWeH + (size_t)l * PK_PER_LAYER,
              be_ih + l * 3 * Hd, be_hh + l * 3 * Hd);
        else
          gru_layer_f32<<<48, 64, 0, stream>>>(
              exin, esrc + l * BH, edst + l * BH,
              We_ih + (size_t)l * 3 * Hd * Hd, We_hh + (size_t)l * 3 * Hd * Hd,
              be_ih + l * 3 * Hd, be_hh + l * 3 * Hd);
        exin  = edst + l * BH;
        exinP = edstPk + l * AP_LAYER;
      }
      head_sigmoid<<<1, 32, 0, stream>>>(edst + 3 * BH, Wf, bf,
                                         out + Bsz * MAXN + i * Bsz * MAXN + j); // edges[i][b][j]
    }
  }
}
